// ARC_conv_43765716746266
// MI455X (gfx1250) — matrix operand order repaired and checked by reading, not compiled, not run on hardware
//
#include <hip/hip_runtime.h>

// ARC glimpse network for MI455X (gfx1250, wave32, WMMA).
// conv2 + LSTM GEMMs run on v_wmma_f32_16x16x32_bf16 (fp32 accumulate).
// Conv biases are dropped exactly: batch-stat BatchNorm is shift-invariant.
// GEMM inner loops batch all fragment loads (ds_load_b128) ahead of a
// back-to-back WMMA burst to avoid per-WMMA dscnt waits and WAR hazard NOPs.

typedef __attribute__((ext_vector_type(16))) __bf16   v16bf;
typedef __attribute__((ext_vector_type(8)))  float    v8f;
typedef __attribute__((ext_vector_type(4)))  unsigned uint4v;

union ABf { v16bf v; uint4v q[2]; unsigned u[8]; };
union CDf { v8f  v; float    f[8]; };

__device__ __forceinline__ unsigned short f2bf(float f) {
  unsigned u = __float_as_uint(f);
  u += 0x7FFFu + ((u >> 16) & 1u);          // round-to-nearest-even
  return (unsigned short)(u >> 16);
}
__device__ __forceinline__ float bf2f(unsigned short h) {
  return __uint_as_float(((unsigned)h) << 16);
}
__device__ __forceinline__ float sigmf(float x) { return 1.0f / (1.0f + __expf(-x)); }

// ---------------------------------------------------------------- init / pack
__global__ void k_init(const float* wih, const float* whh, const float* w2,
                       unsigned short* wihbf, unsigned short* whhbf,
                       unsigned short* w2bf,
                       float* Hx, float* Cx, unsigned short* hbf) {
  int idx = blockIdx.x * 256 + threadIdx.x;
  if (idx < 512 * 4096) wihbf[idx] = f2bf(wih[idx]);
  if (idx < 512 * 128)  whhbf[idx] = f2bf(whh[idx]);
  if (idx < 64 * 576) {                         // A-layout: k = tap*64 + ci
    int m = idx / 576, k = idx % 576, tap = k >> 6, ci = k & 63;
    w2bf[idx] = f2bf(w2[(m * 64 + ci) * 9 + tap]);
  }
  if (idx < 16384) { Hx[idx] = 0.f; Cx[idx] = 0.f; hbf[idx] = 0; }
}

// ------------------------------------------------- conv1 batch stats (pass 1)
__global__ void k_conv1_stats(const float* in, const float* w1,
                              float* pS, float* pQ) {
  __shared__ float img_s[4096];
  __shared__ float w1s[576];
  __shared__ float red[512];
  int id = blockIdx.x;                 // g*128 + b
  int g = id >> 7, b = id & 127;
  const float* src = in + ((size_t)(b * 2 + g)) * 4096;
  for (int i = threadIdx.x; i < 4096; i += 256) img_s[i] = src[i];
  for (int i = threadIdx.x; i < 576;  i += 256) w1s[i]   = w1[i];
  __syncthreads();
  int c = threadIdx.x >> 2, sub = threadIdx.x & 3;   // 64 ch x 4 slices
  float s = 0.f, q = 0.f;
  for (int i = 0; i < 1024; ++i) {
    int p = sub * 1024 + i;
    int y = p >> 6, x = p & 63;
    float a = 0.f;
    #pragma unroll
    for (int dy = 0; dy < 3; ++dy) {
      int iy = y + dy - 1;
      if (iy < 0 || iy > 63) continue;
      #pragma unroll
      for (int dx = 0; dx < 3; ++dx) {
        int ix = x + dx - 1;
        if (ix < 0 || ix > 63) continue;
        a += w1s[c * 9 + dy * 3 + dx] * img_s[iy * 64 + ix];
      }
    }
    s += a; q += a * a;
  }
  red[threadIdx.x] = s; red[256 + threadIdx.x] = q;
  __syncthreads();
  if (sub == 0) {                       // deterministic fixed-order reduce
    int t = threadIdx.x;
    pS[id * 64 + c] = red[t] + red[t + 1] + red[t + 2] + red[t + 3];
    pQ[id * 64 + c] = red[256 + t] + red[257 + t] + red[258 + t] + red[259 + t];
  }
}

__global__ void k_stats_reduce(const float* pS, const float* pQ, int per_g,
                               float* sum, float* sq) {
  int t = threadIdx.x;                 // 128: (g,c)
  int g = t >> 6, c = t & 63;
  float s = 0.f, q = 0.f;
  for (int j = 0; j < per_g; ++j) {
    s += pS[((size_t)g * per_g + j) * 64 + c];
    q += pQ[((size_t)g * per_g + j) * 64 + c];
  }
  sum[t] = s; sq[t] = q;
}

__global__ void k_bn_finalize(const float* sum, const float* sq,
                              const float* gamma, const float* beta,
                              float* A, float* Bsh) {
  int t = threadIdx.x;                 // 128: (g,c)
  int c = t & 63;
  const float N = 128.0f * 64.0f * 64.0f;
  float mean = sum[t] / N;
  float var  = sq[t] / N - mean * mean;
  float rstd = rsqrtf(var + 1e-5f);
  float a = gamma[c] * rstd;
  A[t] = a; Bsh[t] = beta[c] - mean * a;
}

// ------------- fused conv1+bn1+relu -> conv2 implicit GEMM (WMMA bf16) ------
__global__ void __launch_bounds__(256)
k_conv2_fused(const float* in, const float* w1, const unsigned short* w2bf,
              const float* A1, const float* B1, unsigned short* Fout,
              float* pS, float* pQ) {
  extern __shared__ char smem[];
  float* patch = (float*)smem;                       // 400 f
  float* w1s   = patch + 400;                        // 576 f
  float* A1s   = w1s + 576;                          // 64 f
  float* B1s   = A1s + 64;                           // 64 f
  unsigned short* w2s = (unsigned short*)(B1s + 64); // 64x576 bf16 (A matrix), 16B aligned
  unsigned short* r1  = w2s + 36864;                 // 18x18x64 bf16, 32B aligned
  float* red = (float*)r1;                           // reused post-GEMM: 4096 f

  int id = blockIdx.y;                 // g*128 + b
  int g = id >> 7, bb = id & 127;
  int tile = blockIdx.x;               // 16 spatial tiles of 16x16
  int ty = tile >> 2, tx = tile & 3;
  int tid = threadIdx.x;

  for (int i = tid; i < 576; i += 256) w1s[i] = w1[i];
  for (int i = tid; i < 64;  i += 256) { A1s[i] = A1[g * 64 + i]; B1s[i] = B1[g * 64 + i]; }
  for (int i = tid; i < 4608; i += 256)              // 64x576 bf16 as uint4
    ((uint4v*)w2s)[i] = ((const uint4v*)w2bf)[i];
  const float* src = in + ((size_t)(bb * 2 + g)) * 4096;
  for (int p = tid; p < 400; p += 256) {             // 20x20 zero-padded patch
    int py = p / 20, px = p % 20;
    int iy = ty * 16 + py - 2, ix = tx * 16 + px - 2;
    float v = 0.f;
    if (iy >= 0 && iy < 64 && ix >= 0 && ix < 64) v = src[iy * 64 + ix];
    patch[p] = v;
  }
  __syncthreads();
  for (int p = tid; p < 324; p += 256) {             // r1 = relu(bn1(conv1))
    int qy = p / 18, qx = p % 18;
    for (int c = 0; c < 64; ++c) {
      float a = 0.f;
      #pragma unroll
      for (int dy = 0; dy < 3; ++dy)
        #pragma unroll
        for (int dx = 0; dx < 3; ++dx)
          a += w1s[c * 9 + dy * 3 + dx] * patch[(qy + dy) * 20 + (qx + dx)];
      float v = a * A1s[c] + B1s[c];
      r1[p * 64 + c] = f2bf(v > 0.f ? v : 0.f);
    }
  }
  __syncthreads();

  // GEMM M=64 (out ch) x N=256 (pixels) x K=576 (64ci x 9 taps)
  int wave = tid >> 5, lane = tid & 31;
  int lm = lane & 15, lh = lane >> 4;
  int m0 = (wave & 3) * 16;
  int n0 = (wave >> 2) * 128;
  CDf zero;
  #pragma unroll
  for (int e = 0; e < 8; ++e) zero.f[e] = 0.f;
  CDf acc[8];
  #pragma unroll
  for (int t = 0; t < 8; ++t) acc[t] = zero;

  for (int kk = 0; kk < 576; kk += 32) {
    // A 16x32 bf16: half lh holds K = kk+lh*16 .. +15 (contiguous, matches B)
    ABf afr;
    const unsigned short* arow = w2s + (m0 + lm) * 576 + kk + lh * 16;
    afr.q[0] = *(const uint4v*)(arow);
    afr.q[1] = *(const uint4v*)(arow + 8);
    // B 32x16 bf16: half lh holds K = kk+lh*16 .. +15 (16 contiguous bf16 = 32B)
    int kc = kk + lh * 16;
    int tap = kc >> 6, ci = kc & 63;
    int dy = tap / 3, dx = tap % 3;
    ABf bfr[8];
    #pragma unroll
    for (int t = 0; t < 8; ++t) {
      int col = n0 + t * 16 + lm;
      const unsigned short* brow =
          r1 + (((col >> 4) + dy) * 18 + ((col & 15) + dx)) * 64 + ci;
      bfr[t].q[0] = *(const uint4v*)(brow);
      bfr[t].q[1] = *(const uint4v*)(brow + 8);
    }
    #pragma unroll
    for (int t = 0; t < 8; ++t)                     // back-to-back WMMA burst
      acc[t].v = __builtin_amdgcn_wmma_f32_16x16x32_bf16(
          false, afr.v, false, bfr[t].v, (short)0, acc[t].v, false, false);
  }
  // write conv2 output (bias omitted: cancelled by BN) as bf16
  #pragma unroll
  for (int t = 0; t < 8; ++t) {
    int col = n0 + t * 16 + lm;
    int y = ty * 16 + (col >> 4), x = tx * 16 + (col & 15);
    #pragma unroll
    for (int r = 0; r < 8; ++r) {
      int m = m0 + lh * 8 + r;
      Fout[(((size_t)id * 64 + m) << 12) + y * 64 + x] = f2bf(acc[t].f[r]);
    }
  }
  __syncthreads();   // all waves done with r1 -> reuse as reduction scratch
  #pragma unroll
  for (int r = 0; r < 8; ++r) {
    float s = 0.f, q = 0.f;
    #pragma unroll
    for (int t = 0; t < 8; ++t) { float v = acc[t].f[r]; s += v; q += v * v; }
    int m = m0 + lh * 8 + r;
    int slot = (wave >> 2) * 16 + lm;    // 32 contributors per channel
    red[m * 32 + slot] = s;
    red[2048 + m * 32 + slot] = q;
  }
  __syncthreads();
  if (tid < 64) {                        // deterministic per-block partials
    float s = 0.f, q = 0.f;
    for (int j = 0; j < 32; ++j) { s += red[tid * 32 + j]; q += red[2048 + tid * 32 + j]; }
    int slot = id * 16 + tile;
    pS[(size_t)slot * 64 + tid] = s;
    pQ[(size_t)slot * 64 + tid] = q;
  }
}

// ----------------------------------------- bn2 + residual + relu (in place)
__global__ void k_residual(unsigned short* F, const float* in,
                           const float* A2, const float* B2) {
  size_t idx = (size_t)blockIdx.x * 256 + threadIdx.x;   // 2*128*64*4096
  int yx = (int)(idx & 4095);
  int c  = (int)((idx >> 12) & 63);
  int b  = (int)((idx >> 18) & 127);
  int g  = (int)(idx >> 25);
  float v = bf2f(F[idx]) * A2[g * 64 + c] + B2[g * 64 + c]
          + in[((size_t)(b * 2 + g) << 12) + yx];
  F[idx] = f2bf(v > 0.f ? v : 0.f);
}

// ---------------------------------------------------------- per-turn kernels
__global__ void k_glimpse_params(const float* Hx, const float* gw,
                                 const float* gb, float* gp) {
  int b = threadIdx.x;                 // 128
  for (int r = 0; r < 3; ++r) {
    float s = gb[r];
    for (int k = 0; k < 128; ++k) s += Hx[b * 128 + k] * gw[r * 128 + k];
    gp[b * 3 + r] = tanhf(s);
  }
}

__global__ void k_filterbank(const float* gp, float* Fh, float* Fw) {
  int b = blockIdx.x;
  int t = threadIdx.x;
  if (t >= 16) return;
  int row = t & 7;
  bool isW = t >= 8;
  float center = gp[b * 3 + (isW ? 1 : 0)];
  float delta  = gp[b * 3 + 2];
  float ad = fabsf(delta);
  float centers = 63.0f * (center + 1.0f) * 0.5f;
  float deltas  = 8.0f * (1.0f - ad);            // S/G = 64/8
  float gamma   = __expf(1.0f - 2.0f * ad);
  float mu = centers + deltas * ((float)row - 3.5f);
  float inv_g = 1.0f / gamma;
  const float PI = 3.14159265358979323846f;
  float sum = 0.f;
  for (int i = 0; i < 64; ++i) {
    float u = ((float)i - mu) * inv_g;
    sum += 1.0f / (PI * gamma * (1.0f + u * u));
  }
  float norm = 1.0f / (sum + 1e-4f);
  float* dst = (isW ? Fw : Fh) + b * 512 + row * 64;
  for (int i = 0; i < 64; ++i) {
    float u = ((float)i - mu) * inv_g;
    dst[i] = norm / (PI * gamma * (1.0f + u * u));
  }
}

__global__ void k_glimpse(const unsigned short* F, const float* Fh,
                          const float* Fw, unsigned short* flatbf, int gsel) {
  __shared__ unsigned short Ms[4096];
  __shared__ float Fhs[512], Fws[512], ts[512];
  int c = blockIdx.x, b = blockIdx.y;
  int tid = threadIdx.x;               // 64
  const unsigned short* src = F + (((size_t)(gsel * 128 + b) * 64 + c) << 12);
  for (int i = tid; i < 512; i += 64)
    ((uint4v*)Ms)[i] = ((const uint4v*)src)[i];
  for (int i = tid; i < 512; i += 64) { Fhs[i] = Fh[b * 512 + i]; Fws[i] = Fw[b * 512 + i]; }
  __syncthreads();
  int j = tid;                          // t(gy, j) = sum_i Fh(gy,i) * M(i,j)
  for (int gy = 0; gy < 8; ++gy) {
    float a = 0.f;
    for (int i = 0; i < 64; ++i) a += Fhs[gy * 64 + i] * bf2f(Ms[i * 64 + j]);
    ts[gy * 64 + j] = a;
  }
  __syncthreads();
  int gy = tid >> 3, gx = tid & 7;      // gl(gy,gx) = sum_j t(gy,j) * Fw(gx,j)
  float a = 0.f;
  for (int jj = 0; jj < 64; ++jj) a += ts[gy * 64 + jj] * Fws[gx * 64 + jj];
  flatbf[(size_t)b * 4096 + c * 64 + gy * 8 + gx] = f2bf(a);
}

// ------------------------ LSTM gate GEMM: (128 x 512) = flat*Wih^T + Hx*Whh^T
__global__ void __launch_bounds__(256)
k_lstm_gemm(const unsigned short* flatbf, const unsigned short* hbf,
            const unsigned short* wihbf, const unsigned short* whhbf,
            const float* bih, const float* bhh, float* gates) {
  __shared__ uint4v Bs4[256];                        // 64n x 32k bf16 slab
  unsigned short* Bs = (unsigned short*)Bs4;
  int tid = threadIdx.x;
  int wave = tid >> 5, lane = tid & 31;
  int lm = lane & 15, lh = lane >> 4;
  int m0 = wave * 16;                   // 8 waves cover M=128
  int n0 = blockIdx.x * 64;             // 8 blocks cover N=512
  CDf zero;
  #pragma unroll
  for (int e = 0; e < 8; ++e) zero.f[e] = 0.f;
  CDf acc[4];
  #pragma unroll
  for (int t = 0; t < 4; ++t) acc[t] = zero;

  for (int kk = 0; kk < 4096; kk += 32) {          // K1: flat @ w_ih^T
    // stage slab: one uint4 (8 bf16) per thread; row n0+tid/4, k-chunk tid%4
    Bs4[tid] = *(const uint4v*)(wihbf + ((size_t)(n0 + (tid >> 2))) * 4096
                                + kk + ((tid & 3) << 3));
    if (kk + 32 < 4096)                             // gfx1250 global_prefetch_b8
      __builtin_prefetch(wihbf + ((size_t)(n0 + (tid >> 2))) * 4096
                         + kk + 32 + ((tid & 3) << 3), 0, 1);
    __syncthreads();
    ABf afr;
    const unsigned short* arow = flatbf + (size_t)(m0 + lm) * 4096 + kk + lh * 16;
    afr.q[0] = *(const uint4v*)(arow);
    afr.q[1] = *(const uint4v*)(arow + 8);
    ABf bfr[4];
    #pragma unroll
    for (int t = 0; t < 4; ++t) {
      const unsigned short* brow = Bs + (t * 16 + lm) * 32 + lh * 16;
      bfr[t].q[0] = *(const uint4v*)(brow);
      bfr[t].q[1] = *(const uint4v*)(brow + 8);
    }
    #pragma unroll
    for (int t = 0; t < 4; ++t)                     // back-to-back WMMA burst
      acc[t].v = __builtin_amdgcn_wmma_f32_16x16x32_bf16(
          false, afr.v, false, bfr[t].v, (short)0, acc[t].v, false, false);
    __syncthreads();
  }
  for (int kk = 0; kk < 128; kk += 32) {           // K2: Hx @ w_hh^T
    ABf afr;
    const unsigned short* arow = hbf + (m0 + lm) * 128 + kk + lh * 16;
    afr.q[0] = *(const uint4v*)(arow);
    afr.q[1] = *(const uint4v*)(arow + 8);
    ABf bfr[4];
    #pragma unroll
    for (int t = 0; t < 4; ++t) {
      const unsigned short* brow = whhbf + (n0 + t * 16 + lm) * 128 + kk + lh * 16;
      bfr[t].q[0] = *(const uint4v*)(brow);
      bfr[t].q[1] = *(const uint4v*)(brow + 8);
    }
    #pragma unroll
    for (int t = 0; t < 4; ++t)
      acc[t].v = __builtin_amdgcn_wmma_f32_16x16x32_bf16(
          false, afr.v, false, bfr[t].v, (short)0, acc[t].v, false, false);
  }
  #pragma unroll
  for (int t = 0; t < 4; ++t) {
    int n = n0 + t * 16 + lm;
    float bias = bih[n] + bhh[n];
    #pragma unroll
    for (int r = 0; r < 8; ++r) {
      int m = m0 + lh * 8 + r;
      gates[m * 512 + n] = acc[t].f[r] + bias;
    }
  }
}

__global__ void k_lstm_update(const float* gates, float* Hx, float* Cx,
                              unsigned short* hbf) {
  int idx = blockIdx.x * 256 + threadIdx.x;   // 16384
  int b = idx >> 7, h = idx & 127;
  const float* gr = gates + b * 512;
  float gi = sigmf(gr[h]);
  float gf = sigmf(gr[128 + h]);
  float gg = tanhf(gr[256 + h]);
  float go = sigmf(gr[384 + h]);
  float cn = gf * Cx[idx] + gi * gg;
  Cx[idx] = cn;
  float hn = go * tanhf(cn);
  Hx[idx] = hn;
  hbf[idx] = f2bf(hn);
}

// --------------------------------------------------------------------- launch
extern "C" void kernel_launch(void* const* d_in, const int* in_sizes, int n_in,
                              void* d_out, int out_size, void* d_ws, size_t ws_size,
                              hipStream_t stream) {
  (void)in_sizes; (void)n_in; (void)out_size; (void)ws_size;
  const float* img = (const float*)d_in[0];
  const float* w1  = (const float*)d_in[1];
  // d_in[2] conv1_b: exactly cancelled by batch-stat BatchNorm (shift invariant)
  const float* g1  = (const float*)d_in[3];
  const float* be1 = (const float*)d_in[4];
  const float* w2  = (const float*)d_in[5];
  // d_in[6] conv2_b: exactly cancelled by batch-stat BatchNorm
  const float* g2  = (const float*)d_in[7];
  const float* be2 = (const float*)d_in[8];
  const float* wih = (const float*)d_in[9];
  const float* whh = (const float*)d_in[10];
  const float* bih = (const float*)d_in[11];
  const float* bhh = (const float*)d_in[12];
  const float* gw  = (const float*)d_in[13];
  const float* gb  = (const float*)d_in[14];

  char* ws = (char*)d_ws;
  size_t off = 0;
  auto take = [&](size_t bytes) {
    char* p = ws + off;
    off += (bytes + 255) & ~(size_t)255;
    return p;
  };
  float* sum1 = (float*)take(128 * 4);
  float* sq1  = (float*)take(128 * 4);
  float* sum2 = (float*)take(128 * 4);
  float* sq2  = (float*)take(128 * 4);
  float* A1v  = (float*)take(128 * 4);
  float* B1v  = (float*)take(128 * 4);
  float* A2v  = (float*)take(128 * 4);
  float* B2v  = (float*)take(128 * 4);
  float* pS1  = (float*)take((size_t)256 * 64 * 4);
  float* pQ1  = (float*)take((size_t)256 * 64 * 4);
  float* pS2  = (float*)take((size_t)4096 * 64 * 4);
  float* pQ2  = (float*)take((size_t)4096 * 64 * 4);
  unsigned short* w2bf  = (unsigned short*)take((size_t)36864 * 2);
  unsigned short* wihbf = (unsigned short*)take((size_t)512 * 4096 * 2);
  unsigned short* whhbf = (unsigned short*)take((size_t)512 * 128 * 2);
  unsigned short* Fbuf  = (unsigned short*)take((size_t)2 * 128 * 64 * 4096 * 2);
  float* Fh = (float*)take((size_t)128 * 512 * 4);
  float* Fw = (float*)take((size_t)128 * 512 * 4);
  float* gp = (float*)take(128 * 3 * 4);
  unsigned short* flatbf = (unsigned short*)take((size_t)128 * 4096 * 2);
  float* gates = (float*)take((size_t)128 * 512 * 4);
  float* Hx = (float*)take(16384 * 4);
  float* Cx = (float*)take(16384 * 4);
  unsigned short* hbf = (unsigned short*)take(16384 * 2);

  const size_t conv2_smem = 4416 + 73728 + 41472;   // patch/w1/bn + w2s + r1

  k_init<<<8192, 256, 0, stream>>>(wih, whh, w2, wihbf, whhbf, w2bf, Hx, Cx, hbf);
  k_conv1_stats<<<256, 256, 0, stream>>>(img, w1, pS1, pQ1);
  k_stats_reduce<<<1, 128, 0, stream>>>(pS1, pQ1, 128, sum1, sq1);
  k_bn_finalize<<<1, 128, 0, stream>>>(sum1, sq1, g1, be1, A1v, B1v);
  k_conv2_fused<<<dim3(16, 256), 256, conv2_smem, stream>>>(
      img, w1, w2bf, A1v, B1v, Fbuf, pS2, pQ2);
  k_stats_reduce<<<1, 128, 0, stream>>>(pS2, pQ2, 2048, sum2, sq2);
  k_bn_finalize<<<1, 128, 0, stream>>>(sum2, sq2, g2, be2, A2v, B2v);
  k_residual<<<262144, 256, 0, stream>>>(Fbuf, img, A2v, B2v);

  for (int turn = 0; turn < 16; ++turn) {
    int gsel = (turn & 1) ? 0 : 1;     // odd turn -> support (ch 0), even -> test
    k_glimpse_params<<<1, 128, 0, stream>>>(Hx, gw, gb, gp);
    k_filterbank<<<128, 64, 0, stream>>>(gp, Fh, Fw);
    k_glimpse<<<dim3(64, 128), 64, 0, stream>>>(Fbuf, Fh, Fw, flatbf, gsel);
    k_lstm_gemm<<<8, 256, 0, stream>>>(flatbf, hbf, wihbf, whhbf, bih, bhh, gates);
    k_lstm_update<<<64, 256, 0, stream>>>(gates, Hx, Cx, hbf);
  }
  hipMemcpyAsync(d_out, Hx, 16384 * sizeof(float), hipMemcpyDeviceToDevice, stream);
}
